// Pred_angle_Points_11115375362526
// MI455X (gfx1250) — matrix operand order repaired and checked by reading, not compiled, not run on hardware
//
#include <hip/hip_runtime.h>
#include <hip/hip_bf16.h>

typedef unsigned short USH;
typedef __bf16 bf16_t;
typedef bf16_t v16bf __attribute__((ext_vector_type(16)));
typedef float  v8f   __attribute__((ext_vector_type(8)));

#define NROWS      32768              // B*N
#define RAW_STRIDE 16                 // padded 14 -> 16
#define WCAT_HALFS (2*128*768)        // 196608
#define WBLK_HALFS (2*4*128*128)      // 131072
#define WOUT_HALFS (2*16*128)        // 4096

// ---- CDNA5 async global->LDS path (guarded; falls back to sync copy) ----------
#if defined(__has_builtin)
#if __has_builtin(__builtin_amdgcn_global_load_async_to_lds_b128)
#define HAVE_ASYNC_LDS 1
#endif
#endif

typedef int v4i_t __attribute__((ext_vector_type(4)));
typedef __attribute__((address_space(1))) v4i_t* g_v4i_p;   // global
typedef __attribute__((address_space(3))) v4i_t* l_v4i_p;   // LDS

__device__ __forceinline__ void async_copy16(const void* g, void* l) {
#ifdef HAVE_ASYNC_LDS
    __builtin_amdgcn_global_load_async_to_lds_b128((g_v4i_p)g, (l_v4i_p)l, 0, 0);
#else
    *(uint4*)l = *(const uint4*)g;
#endif
}
__device__ __forceinline__ void async_wait() {
#ifdef HAVE_ASYNC_LDS
#if __has_builtin(__builtin_amdgcn_s_wait_asynccnt)
    __builtin_amdgcn_s_wait_asynccnt(0);
#else
    asm volatile("s_wait_asynccnt 0" ::: "memory");
#endif
#endif
}

__device__ __forceinline__ USH f2bf(float f) {
    unsigned int u = __float_as_uint(f);
    unsigned int r = u + 0x7FFFu + ((u >> 16) & 1u);   // RNE
    return (USH)(r >> 16);
}

// ---------------- WMMA fragment helpers (wave32, CDNA5 layouts) ----------------
// A 16x32 bf16: lanes 0-15 row M=lane, K = kb+{0..15} with kb = (lane>=16)?16:0
__device__ __forceinline__ v16bf loadA(const USH* base, int stride, int m, int kofs, int lane) {
    int kb = (lane >> 4) * 16;
    union { unsigned int u[8]; v16bf v; } r;
    const USH* p = base + m * stride + kofs;
#pragma unroll
    for (int j = 0; j < 8; ++j) r.u[j]     = *(const unsigned int*)(p + kb + 2*j);
    return r.v;
}
// B 32x16 bf16 stored transposed in LDS as [N][K]: lane column N=lane&15, K = kb+{0..15}, kb=(lane>=16)?16:0
__device__ __forceinline__ v16bf loadB(const USH* base, int stride, int n, int kofs, int lane) {
    int kb = (lane >> 4) * 16;
    union { unsigned int u[8]; v16bf v; } r;
    const USH* p = base + n * stride + kofs + kb;
#pragma unroll
    for (int j = 0; j < 8; ++j) r.u[j] = *(const unsigned int*)(p + 2*j);
    return r.v;
}
__device__ __forceinline__ v8f wmma_bf16(v16bf a, v16bf b, v8f c) {
    return __builtin_amdgcn_wmma_f32_16x16x32_bf16(false, a, false, b, (short)0, c, false, false);
}

// ---------------- Kernel 0: weight prep (fp32 -> bf16, transposed [N][K]) -------
struct PrepArgs {
    const float* win[2];     // [384][128]
    const float* winit[2];   // [384][128]
    const float* wblk[8];    // [t*4+l], each [128][128] (l: b0.W1,b0.W2,b1.W1,b1.W2)
    const float* wout[2];    // [128][14]
    USH* out;                // WcatT | WblkT | WoutT
};

__global__ __launch_bounds__(256) void prep_weights_kernel(PrepArgs a) {
    int id = blockIdx.x * 256 + threadIdx.x;
    const int total = WCAT_HALFS + WBLK_HALFS + WOUT_HALFS;
    if (id >= total) return;
    float v;
    if (id < WCAT_HALFS) {                       // WcatT[t][n][768]: k<384 from W_in, else W_init
        int t = id / 98304, rem = id % 98304;
        int n = rem / 768, k = rem % 768;
        v = (k < 384) ? a.win[t][k * 128 + n] : a.winit[t][(k - 384) * 128 + n];
    } else if (id < WCAT_HALFS + WBLK_HALFS) {   // WblkT[t][l][n][128]
        int rem = id - WCAT_HALFS;
        int t = rem / 65536; rem %= 65536;
        int l = rem / 16384; rem %= 16384;
        int n = rem / 128, k = rem % 128;
        v = a.wblk[t * 4 + l][k * 128 + n];
    } else {                                     // WoutT[t][16][128], rows 14,15 zero-padded
        int rem = id - WCAT_HALFS - WBLK_HALFS;
        int t = rem / 2048; rem %= 2048;
        int n = rem / 128, k = rem % 128;
        v = (n < 14) ? a.wout[t][k * 14 + n] : 0.f;
    }
    a.out[id] = f2bf(v);
}

// ---------------- Kernel 1: fused resnet trunk (both heads via blockIdx.y) ------
struct GemmArgs {
    const float* s_init;
    const float* s;
    const USH* WcatT;        // [2][128][768]
    const USH* WblkT;        // [2][4][128][128]
    const USH* WoutT;        // [2][16][128]
    const float* b_in[2];
    const float* b_init[2];
    const float* b_blk[8];   // [t*4 + l]
    const float* b_out[2];
    float* raw;              // [2][NROWS][16]
};

__device__ __forceinline__ void gemm128(const USH* A, const USH* Wt, int wm, int wn,
                                        int lane, v8f acc[2][4]) {
#pragma unroll
    for (int ks = 0; ks < 4; ++ks) {
        int ko = ks * 32;
        v16bf a0 = loadA(A, 128, wm * 32 + (lane & 15),      ko, lane);
        v16bf a1 = loadA(A, 128, wm * 32 + 16 + (lane & 15), ko, lane);
#pragma unroll
        for (int nt = 0; nt < 4; ++nt) {
            v16bf b = loadB(Wt, 128, wn * 64 + nt * 16 + (lane & 15), ko, lane);
            acc[0][nt] = wmma_bf16(a0, b, acc[0][nt]);
            acc[1][nt] = wmma_bf16(a1, b, acc[1][nt]);
        }
    }
}

__device__ __forceinline__ void store_relu_bias(USH* Xs, const v8f acc[2][4], int wm, int wn,
                                                int lane, const float* bias) {
    int mo = (lane & 16) ? 8 : 0;
    int n0 = lane & 15;
#pragma unroll
    for (int nt = 0; nt < 4; ++nt) {
        int nn = wn * 64 + nt * 16 + n0;
        float bv = bias ? bias[nn] : 0.f;
#pragma unroll
        for (int mt = 0; mt < 2; ++mt) {
            int mb = wm * 32 + mt * 16 + mo;
#pragma unroll
            for (int r = 0; r < 8; ++r)
                Xs[(mb + r) * 128 + nn] = f2bf(fmaxf(acc[mt][nt][r] + bv, 0.f));
        }
    }
}

// 32KB weight tile, flat 16B-granule copy via async-to-LDS when available
__device__ __forceinline__ void copy_w128(USH* Ws, const USH* src, int tid) {
#pragma unroll
    for (int i = 0; i < 8; ++i) {
        int q = tid + 256 * i;
        async_copy16(src + (size_t)q * 8, Ws + (size_t)q * 8);
    }
}

__global__ __launch_bounds__(256) void trunk_kernel(GemmArgs args) {
    extern __shared__ char smem[];
    USH* As = (USH*)smem;          // [128][32]  layer-1 activation chunk
    USH* Bs = As + 128 * 32;       // [128][32]  layer-1 weight chunk ([N][Kchunk])
    USH* Xs = Bs + 128 * 32;       // [128][128] activation tile
    USH* Ws = Xs + 128 * 128;      // [128][128] weight tile ([N][K])

    const int tid  = threadIdx.x;
    const int lane = tid & 31;
    const int wid  = tid >> 5;
    const int wm   = wid & 3;      // 4 row-groups of 32
    const int wn   = wid >> 2;     // 2 col-groups of 64
    const int t    = blockIdx.y;
    const int row0 = blockIdx.x * 128;

    const USH* WcatT = args.WcatT + (size_t)t * 98304;

    v8f x[2][4];
#pragma unroll
    for (int mt = 0; mt < 2; ++mt)
#pragma unroll
        for (int nt = 0; nt < 4; ++nt) x[mt][nt] = (v8f){0,0,0,0,0,0,0,0};

    // ---- Layer 1: x = relu(s_init)@Win + relu(s)@Winit   (fused K = 768) ----
    // Register-prefetched activation chunks: global loads for chunk ks+1 are issued
    // before the WMMA section of chunk ks so latency overlaps matrix math.
    float4 fA[4];
    {
        const float* src = args.s_init;
#pragma unroll
        for (int i = 0; i < 4; ++i) {
            int j = tid + 256 * i;
            int m = j >> 3, c4 = j & 7;
            fA[i] = ((const float4*)(src + (size_t)(row0 + m) * 384))[c4];
        }
    }
    for (int ks = 0; ks < 24; ++ks) {
        __syncthreads();                               // prior chunk's readers done
#pragma unroll
        for (int i = 0; i < 4; ++i) {                  // store relu+bf16 activations
            int j = tid + 256 * i;
            int m = j >> 3, c4 = j & 7;
            ushort4 h;
            h.x = f2bf(fmaxf(fA[i].x, 0.f)); h.y = f2bf(fmaxf(fA[i].y, 0.f));
            h.z = f2bf(fmaxf(fA[i].z, 0.f)); h.w = f2bf(fmaxf(fA[i].w, 0.f));
            ((ushort4*)(As + m * 32))[c4] = h;
        }
#pragma unroll
        for (int i = 0; i < 2; ++i) {                  // weight chunk [128][32] (async path)
            int q = tid + 256 * i;
            int n = q >> 2, c4 = q & 3;
            async_copy16(WcatT + (size_t)n * 768 + ks * 32 + c4 * 8, Bs + (size_t)q * 8);
        }
        async_wait();
        __syncthreads();

        if (ks + 1 < 24) {                             // prefetch next activation chunk
            const float* src; int kbase;
            if (ks + 1 < 12) { src = args.s_init; kbase = (ks + 1) * 32; }
            else             { src = args.s;      kbase = (ks + 1) * 32 - 384; }
#pragma unroll
            for (int i = 0; i < 4; ++i) {
                int j = tid + 256 * i;
                int m = j >> 3, c4 = j & 7;
                fA[i] = ((const float4*)(src + (size_t)(row0 + m) * 384 + kbase))[c4];
            }
        }

        v16bf a0 = loadA(As, 32, wm * 32 + (lane & 15),      0, lane);
        v16bf a1 = loadA(As, 32, wm * 32 + 16 + (lane & 15), 0, lane);
#pragma unroll
        for (int nt = 0; nt < 4; ++nt) {
            v16bf b = loadB(Bs, 32, wn * 64 + nt * 16 + (lane & 15), 0, lane);
            x[0][nt] = wmma_bf16(a0, b, x[0][nt]);
            x[1][nt] = wmma_bf16(a1, b, x[1][nt]);
        }
    }
#pragma unroll
    for (int nt = 0; nt < 4; ++nt) {                   // + b_in + b_init
        int n = wn * 64 + nt * 16 + (lane & 15);
        float bv = args.b_in[t][n] + args.b_init[t][n];
#pragma unroll
        for (int mt = 0; mt < 2; ++mt)
#pragma unroll
            for (int r = 0; r < 8; ++r) x[mt][nt][r] += bv;
    }

    // ---- ResNet blocks: h = relu(relu(x)@W1+b1); x += h@W2 + b2 ----
    for (int bi = 0; bi < 2; ++bi) {
        __syncthreads();
        store_relu_bias(Xs, x, wm, wn, lane, nullptr);              // relu(x) -> Xs
        copy_w128(Ws, args.WblkT + (size_t)(t * 4 + bi * 2) * 16384, tid);
        async_wait();
        __syncthreads();

        v8f h[2][4];
#pragma unroll
        for (int mt = 0; mt < 2; ++mt)
#pragma unroll
            for (int nt = 0; nt < 4; ++nt) h[mt][nt] = (v8f){0,0,0,0,0,0,0,0};
        gemm128(Xs, Ws, wm, wn, lane, h);

        __syncthreads();
        store_relu_bias(Xs, h, wm, wn, lane, args.b_blk[t * 4 + bi * 2]);  // relu(h+b1) -> Xs
        copy_w128(Ws, args.WblkT + (size_t)(t * 4 + bi * 2 + 1) * 16384, tid);
        async_wait();
        __syncthreads();

        gemm128(Xs, Ws, wm, wn, lane, x);                           // x += h@W2
        const float* b2 = args.b_blk[t * 4 + bi * 2 + 1];
#pragma unroll
        for (int nt = 0; nt < 4; ++nt) {
            float bv = b2[wn * 64 + nt * 16 + (lane & 15)];
#pragma unroll
            for (int mt = 0; mt < 2; ++mt)
#pragma unroll
                for (int r = 0; r < 8; ++r) x[mt][nt][r] += bv;
        }
    }

    // ---- Head: raw = relu(x) @ Wout + b_out   (N padded to 16) ----
    __syncthreads();
    store_relu_bias(Xs, x, wm, wn, lane, nullptr);
    if (tid < 128) {
        async_copy16(args.WoutT + (size_t)t * 2048 + (size_t)tid * 16,
                     Ws + (size_t)tid * 16);
    }
    async_wait();
    __syncthreads();

    v8f o = (v8f){0,0,0,0,0,0,0,0};
#pragma unroll
    for (int ks = 0; ks < 4; ++ks) {
        v16bf a = loadA(Xs, 128, wid * 16 + (lane & 15), ks * 32, lane);
        v16bf b = loadB(Ws, 128, (lane & 15),            ks * 32, lane);
        o = wmma_bf16(a, b, o);
    }
    int n  = lane & 15;
    float bo = (n < 14) ? args.b_out[t][n] : 0.f;
    int mo = (lane & 16) ? 8 : 0;
#pragma unroll
    for (int r = 0; r < 8; ++r) {
        int row = row0 + wid * 16 + r + mo;
        args.raw[((size_t)t * NROWS + row) * RAW_STRIDE + n] = o[r] + bo;
    }
}

// ---------------- Kernel 2: per-residue geometry --------------------------------
__device__ __forceinline__ void m3mul(float* o, const float* a, const float* b) {
#pragma unroll
    for (int i = 0; i < 3; ++i)
#pragma unroll
        for (int j = 0; j < 3; ++j)
            o[i*3+j] = a[i*3+0]*b[0+j] + a[i*3+1]*b[3+j] + a[i*3+2]*b[6+j];
}
__device__ __forceinline__ void m3v(float* o, const float* a, const float* v) {
#pragma unroll
    for (int i = 0; i < 3; ++i)
        o[i] = a[i*3+0]*v[0] + a[i*3+1]*v[1] + a[i*3+2]*v[2];
}

__global__ __launch_bounds__(256) void geometry_kernel(
    const float* __restrict__ rawAr, const float* __restrict__ rawLit,
    const float* __restrict__ rot_mats, const float* __restrict__ trans,
    const int*   __restrict__ aatype,
    const float* __restrict__ lit_positions, const float* __restrict__ default_frames,
    const int*   __restrict__ group_idx, const float* __restrict__ atom_mask,
    float* __restrict__ unnorm, float* __restrict__ angles, float* __restrict__ pred,
    float* __restrict__ rot_g_out, float* __restrict__ tr_g_out)
{
    int idx = blockIdx.x * blockDim.x + threadIdx.x;
    if (idx >= NROWS) return;

    float raw[14], ang[14];
#pragma unroll
    for (int i = 0; i < 14; ++i) raw[i] = rawAr[idx * RAW_STRIDE + i];
#pragma unroll
    for (int a = 0; a < 7; ++a) {
        float s0 = raw[2*a], c0 = raw[2*a+1];
        float inv = rsqrtf(fmaxf(s0*s0 + c0*c0, 1e-8f));
        ang[2*a] = s0 * inv; ang[2*a+1] = c0 * inv;
    }
#pragma unroll
    for (int i = 0; i < 14; ++i) { unnorm[idx*14 + i] = raw[i]; angles[idx*14 + i] = ang[i]; }

    int aa = aatype[idx];

    // per-group frames: rot[g] = def_rot @ x_rot(angle), tr[g] = def_tr
    float rot[8][9], tr[8][3];
#pragma unroll
    for (int g = 0; g < 8; ++g) {
        const float* df = default_frames + (size_t)(aa * 8 + g) * 16;
        float s_ = (g == 0) ? 0.f : ang[2*(g-1)];
        float c_ = (g == 0) ? 1.f : ang[2*(g-1)+1];
#pragma unroll
        for (int i = 0; i < 3; ++i) {
            rot[g][i*3+0] =  df[i*4+0];
            rot[g][i*3+1] =  df[i*4+1]*c_ + df[i*4+2]*s_;
            rot[g][i*3+2] = -df[i*4+1]*s_ + df[i*4+2]*c_;
            tr[g][i]      =  df[i*4+3];
        }
    }
    // compose chi chain (groups 5..7)
    float rbb[8][9], tbb[8][3];
#pragma unroll
    for (int g = 0; g < 5; ++g) {
#pragma unroll
        for (int i = 0; i < 9; ++i) rbb[g][i] = rot[g][i];
#pragma unroll
        for (int i = 0; i < 3; ++i) tbb[g][i] = tr[g][i];
    }
    m3mul(rbb[5], rot[4], rot[5]); m3v(tbb[5], rot[4], tr[5]);
#pragma unroll
    for (int i = 0; i < 3; ++i) tbb[5][i] += tr[4][i];
    m3mul(rbb[6], rbb[5], rot[6]); m3v(tbb[6], rbb[5], tr[6]);
#pragma unroll
    for (int i = 0; i < 3; ++i) tbb[6][i] += tbb[5][i];
    m3mul(rbb[7], rbb[6], rot[7]); m3v(tbb[7], rbb[6], tr[7]);
#pragma unroll
    for (int i = 0; i < 3; ++i) tbb[7][i] += tbb[6][i];

    // apply backbone frame
    float R[9], bt[3];
#pragma unroll
    for (int i = 0; i < 9; ++i) R[i] = rot_mats[idx*9 + i];
#pragma unroll
    for (int i = 0; i < 3; ++i) bt[i] = trans[idx*3 + i] * 10.0f;

    float rg[8][9], tg[8][3];
#pragma unroll
    for (int g = 0; g < 8; ++g) {
        m3mul(rg[g], R, rbb[g]);
        m3v(tg[g], R, tbb[g]);
#pragma unroll
        for (int i = 0; i < 3; ++i) tg[g][i] += bt[i];
#pragma unroll
        for (int i = 0; i < 9; ++i) rot_g_out[(size_t)idx*72 + g*9 + i] = rg[g][i];
#pragma unroll
        for (int i = 0; i < 3; ++i) tr_g_out[(size_t)idx*24 + g*3 + i] = tg[g][i];
    }

    // atom positions
#pragma unroll
    for (int j = 0; j < 14; ++j) {
        float ls = rawLit[idx * RAW_STRIDE + j];
        int g = group_idx[aa*14 + j];
        float lx = lit_positions[(aa*14 + j)*3 + 0] * ls;
        float ly = lit_positions[(aa*14 + j)*3 + 1] * ls;
        float lz = lit_positions[(aa*14 + j)*3 + 2] * ls;
        float mk = atom_mask[aa*14 + j];
#pragma unroll
        for (int i = 0; i < 3; ++i) {
            float p = rg[g][i*3+0]*lx + rg[g][i*3+1]*ly + rg[g][i*3+2]*lz + tg[g][i];
            pred[(size_t)idx*42 + j*3 + i] = p * mk;
        }
    }
}

// ---------------- host launch ----------------------------------------------------
extern "C" void kernel_launch(void* const* d_in, const int* in_sizes, int n_in,
                              void* d_out, int out_size, void* d_ws, size_t ws_size,
                              hipStream_t stream) {
    (void)in_sizes; (void)n_in; (void)out_size;
    const size_t need = (size_t)(WCAT_HALFS + WBLK_HALFS + WOUT_HALFS) * 2
                      + (size_t)2 * NROWS * RAW_STRIDE * 4;
    if (ws_size < need) return;

    // flattened input indices (jax tree order: sorted dict keys, tuples in order)
    const float* s_init   = (const float*)d_in[0];
    const float* s        = (const float*)d_in[1];
    const float* rot_mats = (const float*)d_in[2];
    const float* trans    = (const float*)d_in[3];
    const int*   aatype   = (const int*)  d_in[4];
    // ar: 5..18 ; lit: 19..32
    const float* lit_positions  = (const float*)d_in[33];
    const float* atom_mask      = (const float*)d_in[34];
    const float* default_frames = (const float*)d_in[35];
    const int*   group_idx      = (const int*)  d_in[36];

    USH* wcat = (USH*)d_ws;
    USH* wblk = wcat + WCAT_HALFS;
    USH* wout = wblk + WBLK_HALFS;
    float* raw = (float*)(wout + WOUT_HALFS);        // [2][NROWS][16]

    PrepArgs pa;
    pa.win[0]   = (const float*)d_in[13]; pa.win[1]   = (const float*)d_in[27];
    pa.winit[0] = (const float*)d_in[15]; pa.winit[1] = (const float*)d_in[29];
    pa.wblk[0] = (const float*)d_in[5];  pa.wblk[1] = (const float*)d_in[7];
    pa.wblk[2] = (const float*)d_in[9];  pa.wblk[3] = (const float*)d_in[11];
    pa.wblk[4] = (const float*)d_in[19]; pa.wblk[5] = (const float*)d_in[21];
    pa.wblk[6] = (const float*)d_in[23]; pa.wblk[7] = (const float*)d_in[25];
    pa.wout[0] = (const float*)d_in[17]; pa.wout[1] = (const float*)d_in[31];
    pa.out = wcat;
    {
        int total = WCAT_HALFS + WBLK_HALFS + WOUT_HALFS;
        hipLaunchKernelGGL(prep_weights_kernel, dim3((total + 255) / 256), dim3(256), 0, stream, pa);
    }

    GemmArgs ga;
    ga.s_init = s_init; ga.s = s;
    ga.WcatT = wcat; ga.WblkT = wblk; ga.WoutT = wout;
    ga.b_in[0]   = (const float*)d_in[14]; ga.b_in[1]   = (const float*)d_in[28];
    ga.b_init[0] = (const float*)d_in[16]; ga.b_init[1] = (const float*)d_in[30];
    ga.b_blk[0] = (const float*)d_in[6];  ga.b_blk[1] = (const float*)d_in[8];
    ga.b_blk[2] = (const float*)d_in[10]; ga.b_blk[3] = (const float*)d_in[12];
    ga.b_blk[4] = (const float*)d_in[20]; ga.b_blk[5] = (const float*)d_in[22];
    ga.b_blk[6] = (const float*)d_in[24]; ga.b_blk[7] = (const float*)d_in[26];
    ga.b_out[0] = (const float*)d_in[18]; ga.b_out[1] = (const float*)d_in[32];
    ga.raw = raw;
    {
        size_t lds = (size_t)(128*32 + 128*32 + 128*128 + 128*128) * sizeof(USH); // 80 KB
        hipLaunchKernelGGL(trunk_kernel, dim3(NROWS / 128, 2), dim3(256), lds, stream, ga);
    }

    float* out    = (float*)d_out;
    float* unnorm = out;
    float* angles = out + 458752;           // 32768*14
    float* pred   = out + 917504;
    float* rotg   = out + 2293760;
    float* trg    = out + 4653056;
    const float* rawAr  = raw;
    const float* rawLit = raw + (size_t)NROWS * RAW_STRIDE;

    hipLaunchKernelGGL(geometry_kernel, dim3(NROWS / 256), dim3(256), 0, stream,
                       rawAr, rawLit, rot_mats, trans, aatype,
                       lit_positions, default_frames, group_idx, atom_mask,
                       unnorm, angles, pred, rotg, trg);
}
